// EnhancedHybridAttention_15418932593310
// MI455X (gfx1250) — matrix operand order repaired and checked by reading, not compiled, not run on hardware
//
#include <hip/hip_runtime.h>
#include <cstdint>
#include <cstddef>

// ---------------------------------------------------------------------------
// Types for CDNA5 WMMA (wave32, 16x16x32 bf16 -> f32)
// ---------------------------------------------------------------------------
typedef __attribute__((ext_vector_type(16))) __bf16 v16bf;
typedef __attribute__((ext_vector_type(8)))  float  v8f;

#define SEQ   192
#define DMODEL 1024
#define NHEAD 16
#define DHEAD 64
#define MROWS (SEQ * SEQ)          // 36864 tokens per module

__device__ __forceinline__ unsigned short f2bf_u16(float f) {
  unsigned u = __builtin_bit_cast(unsigned, f);
  u += 0x7FFFu + ((u >> 16) & 1u);            // round-to-nearest-even
  return (unsigned short)(u >> 16);
}
__device__ __forceinline__ __bf16 f2bf(float f) {
  unsigned short s = f2bf_u16(f);
  return __builtin_bit_cast(__bf16, s);
}

__device__ __forceinline__ v8f wmma_bf16(v16bf a, v16bf b, v8f c) {
  return __builtin_amdgcn_wmma_f32_16x16x32_bf16(false, a, false, b, (short)0, c,
                                                 false, false);
}

// A fragment 16x32 (MxK), per ISA table: lanes 0-15 row M=lane, K {0..7,16..23};
// lanes 16-31 same rows, K {8..15,24..31}; 2 bf16 per VGPR.
__device__ __forceinline__ v16bf load_A(const __bf16* p0, int lda, int lane) {
  const int row = lane & 15;
  const int kh  = (lane >> 4) & 1;
  const __bf16* p = p0 + row * lda;
  v16bf a;
#pragma unroll
  for (int i = 0; i < 4; ++i) {
    int k  = kh * 8 + 2 * i;
    a[2 * i]     = p[k];
    a[2 * i + 1] = p[k + 1];
    int k2 = 16 + kh * 8 + 2 * i;
    a[8 + 2 * i] = p[k2];
    a[9 + 2 * i] = p[k2 + 1];
  }
  return a;
}

// B fragment 32x16 (KxN) where memory holds B transposed: lds[n*ldn + k].
// Lanes 0-15: col N=lane, K {0..7,16..23}; lanes 16-31: same cols, K {8..15,24..31}.
__device__ __forceinline__ v16bf load_BT(const __bf16* p0, int ldn, int lane) {
  const __bf16* p = p0 + (lane & 15) * ldn + ((lane >> 4) & 1) * 8;
  v16bf b;
#pragma unroll
  for (int i = 0; i < 8; ++i) { b[i] = p[i]; b[8 + i] = p[16 + i]; }
  return b;
}

// ---------------------------------------------------------------------------
// Kernel 1: dual LayerNorm + transpose + bf16 convert + xl token norms.
// Block = one token (b,s). x is [B,S,d]; xl/xg stored [S,B,d]; xbf [B,S,d].
// LN: unbiased std (ddof=1), eps added to std.
// ---------------------------------------------------------------------------
__global__ void __launch_bounds__(256)
ln_kernel(const float* __restrict__ x,
          const float* __restrict__ gl, const float* __restrict__ bl,
          const float* __restrict__ gg, const float* __restrict__ bg,
          __bf16* __restrict__ xl, __bf16* __restrict__ xg,
          __bf16* __restrict__ xbf, float* __restrict__ norms) {
  const int token = blockIdx.x;            // b*SEQ + s
  const int b = token / SEQ, s = token % SEQ;
  const float* row = x + (size_t)token * DMODEL;
  const int t = threadIdx.x;
  __shared__ float r1[256], r2[256];

  float v4[4], s1 = 0.f, s2 = 0.f;
#pragma unroll
  for (int i = 0; i < 4; ++i) {
    float v = row[t + i * 256];
    v4[i] = v; s1 += v; s2 += v * v;
  }
  r1[t] = s1; r2[t] = s2;
  __syncthreads();
  for (int o = 128; o > 0; o >>= 1) {
    if (t < o) { r1[t] += r1[t + o]; r2[t] += r2[t + o]; }
    __syncthreads();
  }
  const float mean = r1[0] * (1.f / DMODEL);
  const float var  = (r2[0] - (float)DMODEL * mean * mean) * (1.f / (DMODEL - 1));
  const float inv  = 1.f / (sqrtf(fmaxf(var, 0.f)) + 1e-6f);
  __syncthreads();

  const size_t otl = ((size_t)(s * SEQ + b)) * DMODEL;
  float nsq = 0.f;
#pragma unroll
  for (int i = 0; i < 4; ++i) {
    int c = t + i * 256;
    float y  = (v4[i] - mean) * inv;
    float a  = gl[c] * y + bl[c];
    float g2 = gg[c] * y + bg[c];
    xl[otl + c] = f2bf(a);
    xg[otl + c] = f2bf(g2);
    xbf[(size_t)token * DMODEL + c] = f2bf(v4[i]);
    nsq += a * a;
  }
  r1[t] = nsq;
  __syncthreads();
  for (int o = 128; o > 0; o >>= 1) {
    if (t < o) r1[t] += r1[t + o];
    __syncthreads();
  }
  if (t == 0) norms[s * SEQ + b] = sqrtf(r1[0]);
}

// ---------------------------------------------------------------------------
// Kernel 2: per-position descending argsort of 192 norms (== top_k with k=192;
// ties broken by lower index first, matching jax.lax.top_k).
// ---------------------------------------------------------------------------
__global__ void __launch_bounds__(192)
argsort_kernel(const float* __restrict__ norms, int* __restrict__ idx) {
  const int s = blockIdx.x;
  const int b = threadIdx.x;
  __shared__ float vals[SEQ];
  vals[b] = norms[s * SEQ + b];
  __syncthreads();
  const float v = vals[b];
  int rank = 0;
  for (int j = 0; j < SEQ; ++j) {
    float vj = vals[j];
    rank += (vj > v) || (vj == v && j < b);
  }
  idx[s * SEQ + rank] = b;
}

// ---------------------------------------------------------------------------
// Kernel 2b: fp32 -> bf16 weight conversion (pairwise).
// ---------------------------------------------------------------------------
__global__ void __launch_bounds__(256)
cvt_kernel(const float* __restrict__ s, __bf16* __restrict__ d, int npairs) {
  int i = blockIdx.x * 256 + threadIdx.x;
  if (i < npairs) {
    float2 f = ((const float2*)s)[i];
    unsigned p = (unsigned)f2bf_u16(f.x) | ((unsigned)f2bf_u16(f.y) << 16);
    ((unsigned*)d)[i] = p;
  }
}

// ---------------------------------------------------------------------------
// Kernel 3: tiled bf16 WMMA GEMM  C[M,N] = A[M,1024] * Wb[N,1024]^T + bias.
// 256 threads (8 waves), 128x128 tile, K-step 32. Wave w -> rows 32*(w>>1),
// cols 64*(w&1): 2x4 fragment grid; all 4 B frags preloaded, then an
// 8-WMMA burst so ds_loads pipeline against the previous burst.
// mode 0: store bf16 to qkvout[r*Ncols+col]
// mode 1: out[(n*192+l)*1024+col]  = val  (l=r/192, n=r%192)  -- first module
// mode 2: out[(n*192+l)*1024+col] += val                       -- accumulate
// ---------------------------------------------------------------------------
__global__ void __launch_bounds__(256)
gemm_bf16(const __bf16* __restrict__ A, const __bf16* __restrict__ Wb,
          const float* __restrict__ bias, __bf16* __restrict__ qkvout,
          float* __restrict__ accout, int Ncols, int mode) {
  const int t = threadIdx.x;
  const int lane = t & 31;
  const int w = t >> 5;
  const int m0 = blockIdx.y * 128;
  const int n0 = blockIdx.x * 128;

  __shared__ __bf16 Atile[128 * 40];   // 128 rows x 32 k (+8 pad)
  __shared__ __bf16 Btile[128 * 40];   // 128 ncols x 32 k (+8 pad)

  v8f acc[2][4] = {{{}, {}, {}, {}}, {{}, {}, {}, {}}};
  const int wr = w >> 1;               // 0..3  -> 32-row slice
  const int wc = w & 1;                // 0..1  -> 64-col slice

  const uint32_t* Ag = (const uint32_t*)A;
  const uint32_t* Bg = (const uint32_t*)Wb;

  for (int kt = 0; kt < DMODEL; kt += 32) {
    if (kt + 32 < DMODEL) {  // WGP-scope prefetch of next K-tile
      __builtin_prefetch(A  + ((size_t)(m0 + (t & 127)) * DMODEL + kt + 32), 0, 3);
      __builtin_prefetch(Wb + ((size_t)(n0 + (t & 127)) * DMODEL + kt + 32), 0, 3);
    }
#pragma unroll
    for (int i = 0; i < 8; ++i) {
      int pe = t + i * 256;                    // 0..2047 uint pairs
      int r = pe >> 4, c2 = pe & 15;
      *(uint32_t*)&Atile[r * 40 + c2 * 2] =
          Ag[(((size_t)(m0 + r)) * DMODEL + kt) / 2 + c2];
      *(uint32_t*)&Btile[r * 40 + c2 * 2] =
          Bg[(((size_t)(n0 + r)) * DMODEL + kt) / 2 + c2];
    }
    __syncthreads();

    v16bf a0 = load_A(Atile + (32 * wr) * 40, 40, lane);
    v16bf a1 = load_A(Atile + (32 * wr + 16) * 40, 40, lane);
    v16bf bfr[4];
#pragma unroll
    for (int fj = 0; fj < 4; ++fj)
      bfr[fj] = load_BT(Btile + (64 * wc + 16 * fj) * 40, 40, lane);
#pragma unroll
    for (int fj = 0; fj < 4; ++fj) {
      acc[0][fj] = wmma_bf16(a0, bfr[fj], acc[0][fj]);
      acc[1][fj] = wmma_bf16(a1, bfr[fj], acc[1][fj]);
    }
    __syncthreads();
  }

#pragma unroll
  for (int fi = 0; fi < 2; ++fi)
#pragma unroll
    for (int fj = 0; fj < 4; ++fj) {
      int col = n0 + 64 * wc + 16 * fj + (lane & 15);
      float bv = bias[col];
      int rbase = m0 + 32 * wr + 16 * fi + ((lane >> 4) << 3);
      if (mode == 0) {
#pragma unroll
        for (int i = 0; i < 8; ++i)
          qkvout[(size_t)(rbase + i) * Ncols + col] = f2bf(acc[fi][fj][i] + bv);
      } else {
        // rbase is a multiple of 8 and 192 % 8 == 0, so the 8-row span never
        // crosses a multiple of 192: one div/mod per fragment.
        int l = rbase / SEQ, nn = rbase % SEQ;
        float* p0 = accout + ((size_t)(nn * SEQ + l)) * DMODEL + col;
#pragma unroll
        for (int i = 0; i < 8; ++i) {
          float v = acc[fi][fj][i] + bv;
          float* p = p0 + (size_t)i * SEQ * DMODEL;
          if (mode == 1) *p = v; else *p += v;
        }
      }
    }
}

// ---------------------------------------------------------------------------
// Kernel 4: attention for one (batch column n, head h). 192 threads = 6 waves.
// QKV rows gathered via gidx for the sparse module (gidx==nullptr otherwise).
// Dynamic LDS: Q[192x64] + K[192x64] + V^T[64x192] bf16 + per-wave P strip
// [16x192] bf16 = 108 KB. V is staged transposed so P x V value fragments are
// contiguous load_BT (2 x ds_load_b128) instead of 16 strided ds_load_u16.
// ---------------------------------------------------------------------------
#define ATTN_SMEM ((3 * SEQ * DHEAD + 6 * 16 * SEQ) * 2)

__global__ void __launch_bounds__(192)
attn_kernel(const __bf16* __restrict__ qkv, const int* __restrict__ gidx,
            __bf16* __restrict__ ctx) {
  const int t = threadIdx.x;
  const int lane = t & 31;
  const int w = t >> 5;
  const int n = blockIdx.x % SEQ;
  const int h = blockIdx.x / SEQ;

  extern __shared__ char smem[];
  __bf16* Qs = (__bf16*)smem;
  __bf16* Ks = Qs + SEQ * DHEAD;
  __bf16* Vt = Ks + SEQ * DHEAD;                 // [DHEAD][SEQ]
  __bf16* Ps = Vt + DHEAD * SEQ + w * (16 * SEQ);

  {  // stage Q,K (natural) and V (transposed) for this (n,h)
    int l = t;
    int nn = gidx ? gidx[l * SEQ + n] : n;
    size_t rb = ((size_t)(l * SEQ + nn)) * (3 * DMODEL) + h * DHEAD;
    const float4* sq = (const float4*)(qkv + rb);
    const float4* sk = (const float4*)(qkv + rb + DMODEL);
    const float4* sv = (const float4*)(qkv + rb + 2 * DMODEL);
    float4* dq = (float4*)(Qs + l * DHEAD);
    float4* dk = (float4*)(Ks + l * DHEAD);
#pragma unroll
    for (int i = 0; i < 8; ++i) {
      dq[i] = sq[i];
      dk[i] = sk[i];
      float4 fv = sv[i];
      const __bf16* vp = (const __bf16*)&fv;     // 8 bf16 per float4
#pragma unroll
      for (int j = 0; j < 8; ++j) Vt[(i * 8 + j) * SEQ + l] = vp[j];
    }
  }
  __syncthreads();

  const float scale = 0.125f;  // 1/sqrt(64)
  const int c = lane & 15;
  const int g = (lane >> 4) & 1;

  for (int it = 0; it < 2; ++it) {       // 12 strips / 6 waves
    const int row0 = (w + 6 * it) * 16;
    v16bf qa0 = load_A(Qs + row0 * DHEAD, DHEAD, lane);
    v16bf qa1 = load_A(Qs + row0 * DHEAD + 32, DHEAD, lane);

    v8f sc[12];
#pragma unroll
    for (int mt = 0; mt < 12; ++mt) {
      v16bf b0 = load_BT(Ks + mt * 16 * DHEAD, DHEAD, lane);
      v16bf b1 = load_BT(Ks + mt * 16 * DHEAD + 32, DHEAD, lane);
      v8f z = {};
      v8f s0 = wmma_bf16(qa0, b0, z);
      sc[mt] = wmma_bf16(qa1, b1, s0);
    }

    // softmax: each 16-lane half holds one row's 16 cols per tile
#pragma unroll
    for (int i = 0; i < 8; ++i) {
      float m = -3.4e38f;
#pragma unroll
      for (int mt = 0; mt < 12; ++mt) m = fmaxf(m, sc[mt][i]);
      for (int off = 1; off < 16; off <<= 1) m = fmaxf(m, __shfl_xor(m, off, 32));
      m *= scale;
      float s = 0.f;
#pragma unroll
      for (int mt = 0; mt < 12; ++mt) {
        float e = __expf(sc[mt][i] * scale - m);
        sc[mt][i] = e;
        s += e;
      }
      for (int off = 1; off < 16; off <<= 1) s += __shfl_xor(s, off, 32);
      float invs = 1.f / s;
      int rl = g * 8 + i;
#pragma unroll
      for (int mt = 0; mt < 12; ++mt)
        Ps[rl * SEQ + mt * 16 + c] = f2bf(sc[mt][i] * invs);
    }
    __syncthreads();

    // O(16x64) = P(16x192) x V(192x64), value frags from transposed V
    v8f oa[4] = {{}, {}, {}, {}};
#pragma unroll
    for (int ks = 0; ks < 6; ++ks) {
      v16bf pa = load_A(Ps + ks * 32, SEQ, lane);
#pragma unroll
      for (int nt = 0; nt < 4; ++nt) {
        v16bf vb = load_BT(Vt + (nt * 16) * SEQ + ks * 32, SEQ, lane);
        oa[nt] = wmma_bf16(pa, vb, oa[nt]);
      }
    }
#pragma unroll
    for (int nt = 0; nt < 4; ++nt)
#pragma unroll
      for (int i = 0; i < 8; ++i) {
        int l = row0 + g * 8 + i;
        ctx[((size_t)(l * SEQ + n)) * DMODEL + h * DHEAD + nt * 16 + c] =
            f2bf(oa[nt][i]);
      }
    __syncthreads();
  }
}

// ---------------------------------------------------------------------------
// Host launcher
// ---------------------------------------------------------------------------
extern "C" void kernel_launch(void* const* d_in, const int* in_sizes, int n_in,
                              void* d_out, int out_size, void* d_ws,
                              size_t ws_size, hipStream_t stream) {
  (void)in_sizes; (void)n_in; (void)out_size; (void)ws_size;
  const float* x = (const float*)d_in[0];
  const float* w_in[4]  = {(const float*)d_in[1],  (const float*)d_in[5],
                           (const float*)d_in[9],  (const float*)d_in[13]};
  const float* b_in[4]  = {(const float*)d_in[2],  (const float*)d_in[6],
                           (const float*)d_in[10], (const float*)d_in[14]};
  const float* w_out[4] = {(const float*)d_in[3],  (const float*)d_in[7],
                           (const float*)d_in[11], (const float*)d_in[15]};
  const float* b_out[4] = {(const float*)d_in[4],  (const float*)d_in[8],
                           (const float*)d_in[12], (const float*)d_in[16]};
  const float* lng = (const float*)d_in[17];
  const float* lnb = (const float*)d_in[18];
  const float* gng = (const float*)d_in[19];
  const float* gnb = (const float*)d_in[20];
  float* out = (float*)d_out;

  char* ws = (char*)d_ws;
  size_t off = 0;
  auto carve = [&](size_t bytes) -> void* {
    void* p = ws + off;
    off = (off + bytes + 255) & ~(size_t)255;
    return p;
  };
  const size_t tokbytes = (size_t)MROWS * DMODEL * 2;
  __bf16* xl  = (__bf16*)carve(tokbytes);
  __bf16* xg  = (__bf16*)carve(tokbytes);
  __bf16* xbf = (__bf16*)carve(tokbytes);
  __bf16* qkv = (__bf16*)carve((size_t)MROWS * 3 * DMODEL * 2);
  __bf16* ctx = (__bf16*)carve(tokbytes);
  __bf16* wbi = (__bf16*)carve((size_t)3 * DMODEL * DMODEL * 2);
  __bf16* wbo = (__bf16*)carve((size_t)DMODEL * DMODEL * 2);
  float*  norms = (float*)carve((size_t)MROWS * 4);
  int*    idx   = (int*)carve((size_t)MROWS * 4);

  ln_kernel<<<MROWS, 256, 0, stream>>>(x, lng, lnb, gng, gnb, xl, xg, xbf, norms);
  argsort_kernel<<<SEQ, SEQ, 0, stream>>>(norms, idx);

  const int np_in  = 3 * DMODEL * DMODEL / 2;
  const int np_out = DMODEL * DMODEL / 2;

  for (int m = 0; m < 4; ++m) {
    const __bf16* inp = (m == 0) ? xl : (m == 1) ? xg : xbf;
    cvt_kernel<<<(np_in + 255) / 256, 256, 0, stream>>>(w_in[m], wbi, np_in);
    // QKV projection: [36864,1024] x [1024,3072]
    gemm_bf16<<<dim3(3 * DMODEL / 128, MROWS / 128), 256, 0, stream>>>(
        inp, wbi, b_in[m], qkv, nullptr, 3 * DMODEL, 0);
    // attention per (n,h)
    attn_kernel<<<SEQ * NHEAD, 192, ATTN_SMEM, stream>>>(
        qkv, (m == 0) ? idx : nullptr, ctx);
    // out projection, transposed scatter-accumulate into d_out
    cvt_kernel<<<(np_out + 255) / 256, 256, 0, stream>>>(w_out[m], wbo, np_out);
    gemm_bf16<<<dim3(DMODEL / 128, MROWS / 128), 256, 0, stream>>>(
        ctx, wbo, b_out[m], nullptr, out, DMODEL, (m == 0) ? 1 : 2);
  }
}
